// PPOTimeLSTMPolicy_10170482557293
// MI455X (gfx1250) — matrix operand order repaired and checked by reading, not compiled, not run on hardware
//
#include <hip/hip_runtime.h>
#include <hip/hip_bf16.h>

typedef _Float16 v16h_t __attribute__((ext_vector_type(16)));
typedef _Float16 h8_t   __attribute__((ext_vector_type(8)));
typedef float    v8f_t  __attribute__((ext_vector_type(8)));

#define B_      64
#define T_      128
#define BT_     8192            // B*T
#define MAXR_   16
#define MR_     131072          // BT*MAXR
#define OBSW_   568
#define TDIDX_  44
#define LSTMH_  512

__device__ __forceinline__ float sigmoidf_(float x) { return 1.0f / (1.0f + __expf(-x)); }

// ---------------------------------------------------------------------------
// Tile staging: 16 f32 -> f16 into LDS (two 16B chunks).
// `full` is BLOCK-UNIFORM (derived from blockIdx + SGPR sizes): fast path is
// straight global_load_b64 at immediate offsets; slow path clamps + selects.
// Requires K even and >= 2 (true for every GEMM in this model).
// ---------------------------------------------------------------------------
__device__ __forceinline__ void stage16(
    const float* __restrict__ P, int ld, int rowLimit, int gr,
    int kbase, int K, _Float16* __restrict__ dst, bool full)
{
    h8_t lo, hi8;
    if (full) {
        const float* p = P + (size_t)gr * (size_t)ld + kbase;
        __builtin_prefetch(p + 32, 0, 3);           // next K-tile -> global_prefetch_b8
        #pragma unroll
        for (int q = 0; q < 8; ++q) {
            float2 v = *(const float2*)(p + q * 2);
            if (q < 4) { lo[2*q]      = (_Float16)v.x; lo[2*q+1]      = (_Float16)v.y; }
            else       { hi8[2*(q-4)] = (_Float16)v.x; hi8[2*(q-4)+1] = (_Float16)v.y; }
        }
    } else {
        const int grc = min(gr, rowLimit - 1);
        const float* p = P + (size_t)grc * (size_t)ld;
        const bool rok = gr < rowLimit;
        #pragma unroll
        for (int q = 0; q < 8; ++q) {
            const int gk  = kbase + q * 2;
            const int gkc = min(gk, K - 2);
            float2 v = *(const float2*)(p + gkc);   // always in-bounds
            const bool ok = rok && (gk < K);        // chunk fully valid or not
            _Float16 x = ok ? (_Float16)v.x : (_Float16)0.0f;
            _Float16 y = ok ? (_Float16)v.y : (_Float16)0.0f;
            if (q < 4) { lo[2*q]      = x; lo[2*q+1]      = y; }
            else       { hi8[2*(q-4)] = x; hi8[2*(q-4)+1] = y; }
        }
    }
    *(h8_t*)(dst)     = lo;
    *(h8_t*)(dst + 8) = hi8;
}

// ---------------------------------------------------------------------------
// Generic WMMA GEMM with optional fused second operand pair:
//   Y[M,N] = act( X[M,K] @ W[N,K]^T  +  X2[M,K2] @ W2[N,K2]^T  + bias )
// 128 threads (4 wave32), 64x64 tile, BK=32, ping-pong LDS (1 barrier/tile),
// f16 WMMA with f32 accumulation. K2==0 disables the second pair.
// ---------------------------------------------------------------------------
__global__ __launch_bounds__(128) void gemm_wmma(
    const float* __restrict__ X,  int ldx,
    const float* __restrict__ X2, int ldx2,
    const float* __restrict__ W,  int ldw,
    const float* __restrict__ W2, int ldw2,
    const float* __restrict__ bias,
    float* __restrict__ Y, int ldy,
    int M, int N, int K, int K2, int act)
{
    __shared__ _Float16 Xs[2][64][40];   // [buf][m][k], 80B row stride
    __shared__ _Float16 Ws[2][64][40];   // [buf][n][k]

    const int tid  = threadIdx.x;
    const int lane = tid & 31;
    const int wave = tid >> 5;
    const int lm   = lane & 15;          // lane-in-halfwave
    const int hi   = lane >> 4;          // which 16-lane half
    const int wr   = (wave >> 1) << 5;   // wave row offset in tile (0/32)
    const int wc   = (wave & 1) << 5;    // wave col offset in tile (0/32)
    const int m0   = blockIdx.x << 6;
    const int n0   = blockIdx.y << 6;

    v8f_t acc[2][2] = {};

    const int sr = tid >> 1;             // staging row (0..63)
    const int sc = (tid & 1) << 4;       // staging col group (0 or 16)

    const int kt1    = (K + 31) >> 5;
    const int kt2    = K2 ? ((K2 + 31) >> 5) : 0;
    const int ktiles = kt1 + kt2;

    const bool rowsFullX = (m0 + 64 <= M);
    const bool rowsFullW = (n0 + 64 <= N);

    // stage tile `ti` (segment-selected) into LDS buffer `buf`
    auto stagePair = [&](int ti, int buf) {
        const float *pX, *pW;
        int lx, lw, kk, kb;
        if (ti < kt1) { pX = X;  pW = W;  lx = ldx;  lw = ldw;  kk = K;  kb = ti << 5; }
        else          { pX = X2; pW = W2; lx = ldx2; lw = ldw2; kk = K2; kb = (ti - kt1) << 5; }
        const bool kfull = (kb + 32 <= kk);
        stage16(pX, lx, M, m0 + sr, kb + sc, kk, &Xs[buf][sr][sc], rowsFullX && kfull);
        stage16(pW, lw, N, n0 + sr, kb + sc, kk, &Ws[buf][sr][sc], rowsFullW && kfull);
    };

    stagePair(0, 0);

    int p = 0;
    for (int ti = 0; ti < ktiles; ++ti) {
        __syncthreads();
        if (ti + 1 < ktiles) stagePair(ti + 1, p ^ 1);

        // ---- fragments per ISA 16-bit A/B layouts ----
        v16h_t a[2], b[2];
        #pragma unroll
        for (int mi = 0; mi < 2; ++mi) {
            const int row = wr + mi * 16 + lm;
            h8_t l0 = *(const h8_t*)&Xs[p][row][hi * 8];        // K = hi*8 .. +7
            h8_t l1 = *(const h8_t*)&Xs[p][row][16 + hi * 8];   // K = 16+hi*8 .. +7
            a[mi] = __builtin_shufflevector(l0, l1, 0,1,2,3,4,5,6,7,8,9,10,11,12,13,14,15);
        }
        #pragma unroll
        for (int ni = 0; ni < 2; ++ni) {
            const int col = wc + ni * 16 + lm;
            h8_t l0 = *(const h8_t*)&Ws[p][col][hi * 8];        // K = hi*8 .. +7
            h8_t l1 = *(const h8_t*)&Ws[p][col][16 + hi * 8];   // K = 16+hi*8 .. +7
            b[ni] = __builtin_shufflevector(l0, l1, 0,1,2,3,4,5,6,7,8,9,10,11,12,13,14,15);
        }
        #pragma unroll
        for (int mi = 0; mi < 2; ++mi)
            #pragma unroll
            for (int ni = 0; ni < 2; ++ni)
                acc[mi][ni] = __builtin_amdgcn_wmma_f32_16x16x32_f16(
                    false, a[mi], false, b[ni], (short)0, acc[mi][ni], false, false);
        p ^= 1;
    }

    // ---- epilogue: bias + activation + store ----
    #pragma unroll
    for (int mi = 0; mi < 2; ++mi) {
        #pragma unroll
        for (int ni = 0; ni < 2; ++ni) {
            const int col = n0 + wc + ni * 16 + lm;
            if (col >= N) continue;
            const float bv = bias ? bias[col] : 0.0f;
            #pragma unroll
            for (int r = 0; r < 8; ++r) {
                const int row = m0 + wr + mi * 16 + hi * 8 + r;
                if (row >= M) continue;
                float v = acc[mi][ni][r] + bv;
                if (act == 1)      v = fmaxf(v, 0.0f);
                else if (act == 2) v = sigmoidf_(v);
                Y[(size_t)row * (size_t)ldy + col] = v;
            }
        }
    }
}

// ---------------------------------------------------------------------------
// obs[b,t, 56 + r*32 + k]  ->  Xr[(bt*16 + r)*32 + k]
// ---------------------------------------------------------------------------
__global__ __launch_bounds__(256) void extract_runners(
    const float* __restrict__ obs, float* __restrict__ Xr)
{
    size_t idx = (size_t)blockIdx.x * 256 + threadIdx.x;
    if (idx >= (size_t)MR_ * 32) return;
    int k = idx & 31;
    size_t m = idx >> 5;
    int r = (int)(m & 15);
    size_t bt = m >> 4;
    Xr[idx] = obs[bt * OBSW_ + 56 + r * 32 + k];
}

// ---------------------------------------------------------------------------
// lstm_in[bt] = concat(m_emb[bt], mean_r r_emb, max_r r_emb)
// ---------------------------------------------------------------------------
__global__ __launch_bounds__(256) void build_lstm_in(
    const float* __restrict__ m_emb, const float* __restrict__ r_emb,
    float* __restrict__ out)
{
    size_t idx = (size_t)blockIdx.x * 256 + threadIdx.x;
    if (idx >= (size_t)BT_ * 256) return;
    int j = idx & 255;
    size_t bt = idx >> 8;
    const float* base = r_emb + (bt * 16) * 256 + j;
    float v0 = base[0];
    float mean = v0, mx = v0;
    #pragma unroll
    for (int r = 1; r < 16; ++r) {
        float v = base[(size_t)r * 256];
        mean += v;
        mx = fmaxf(mx, v);
    }
    mean *= (1.0f / 16.0f);
    float* o = out + bt * 768;
    o[j]       = m_emb[idx];
    o[256 + j] = mean;
    o[512 + j] = mx;
}

// ---------------------------------------------------------------------------
// LSTM gate nonlinearity + state update (one layer, one time step)
// gates[64][2048] = [i f g o]; f gets + Wdt*dt. Writes h,c; opt lstm_out row.
// ---------------------------------------------------------------------------
__global__ __launch_bounds__(256) void lstm_pointwise(
    const float* __restrict__ gates, const float* __restrict__ Wdt,
    const float* __restrict__ obs, int t,
    float* __restrict__ h, float* __restrict__ c,
    float* __restrict__ lstm_out)
{
    int idx = blockIdx.x * 256 + threadIdx.x;
    if (idx >= B_ * LSTMH_) return;
    int b = idx >> 9;
    int j = idx & (LSTMH_ - 1);
    float dt = obs[((size_t)b * T_ + t) * OBSW_ + TDIDX_];
    const float* g = gates + (size_t)b * 2048;
    float iv = sigmoidf_(g[j]);
    float fv = sigmoidf_(g[512 + j] + Wdt[j] * dt);
    float gv = tanhf(g[1024 + j]);
    float ov = sigmoidf_(g[1536 + j]);
    float cn = fv * c[idx] + iv * gv;
    float hn = ov * tanhf(cn);
    c[idx] = cn;
    h[idx] = hn;
    if (lstm_out) lstm_out[((size_t)b * T_ + t) * LSTMH_ + j] = hn;
}

// ---------------------------------------------------------------------------
// hidden = relu(G1 + G2[bt] + fill*W1[:,768] + mature*W1[:,769] + b1) in-place
// ---------------------------------------------------------------------------
__global__ __launch_bounds__(256) void actor_epilogue(
    float* __restrict__ G1, const float* __restrict__ G2,
    const float* __restrict__ fill, const float* __restrict__ mature,
    const float* __restrict__ W1, const float* __restrict__ b1)
{
    size_t idx = (size_t)blockIdx.x * 256 + threadIdx.x;
    if (idx >= (size_t)MR_ * 256) return;
    int n = (int)(idx & 255);
    size_t m = idx >> 8;
    int r = (int)(m & 15);
    size_t bt = m >> 4;
    float v = G1[idx] + G2[bt * 256 + n]
            + fill[bt * 16 + r]   * W1[(size_t)n * 770 + 768]
            + mature[bt * 16 + r] * W1[(size_t)n * 770 + 769]
            + b1[n];
    G1[idx] = fmaxf(v, 0.0f);
}

// ---------------------------------------------------------------------------
// out[bt][193] = [act_mean(64) log_std(64) value(1) fill(16) mature(16)
//                 pnl(16) clip(log_var)(16)]
// ---------------------------------------------------------------------------
__global__ __launch_bounds__(256) void assemble_out(
    const float* __restrict__ act, const float* __restrict__ log_std,
    const float* __restrict__ value, const float* __restrict__ fill,
    const float* __restrict__ mature, const float* __restrict__ risk,
    float* __restrict__ out)
{
    int idx = blockIdx.x * 256 + threadIdx.x;
    if (idx >= BT_ * 193) return;
    int bt = idx / 193;
    int j  = idx - bt * 193;
    float v;
    if (j < 64)        v = act[((size_t)bt * 16 + (j >> 2)) * 4 + (j & 3)];
    else if (j < 128)  v = log_std[j - 64];
    else if (j == 128) v = value[bt];
    else if (j < 145)  v = fill[bt * 16 + (j - 129)];
    else if (j < 161)  v = mature[bt * 16 + (j - 145)];
    else if (j < 177)  v = risk[bt * 32 + (j - 161)];
    else               v = fminf(fmaxf(risk[bt * 32 + 16 + (j - 177)], -8.0f), 4.0f);
    out[idx] = v;
}

// ---------------------------------------------------------------------------
static inline void gemm(hipStream_t s, const float* X, int ldx,
                        const float* W, int ldw, const float* bias,
                        float* Y, int ldy, int M, int N, int K, int act)
{
    dim3 grid((M + 63) / 64, (N + 63) / 64);
    gemm_wmma<<<grid, dim3(128), 0, s>>>(X, ldx, nullptr, 0, W, ldw, nullptr, 0,
                                         bias, Y, ldy, M, N, K, 0, act);
}

static inline void gemm_fused2(hipStream_t s,
                               const float* X,  int ldx,  const float* W,  int ldw,  int K,
                               const float* X2, int ldx2, const float* W2, int ldw2, int K2,
                               const float* bias, float* Y, int ldy, int M, int N, int act)
{
    dim3 grid((M + 63) / 64, (N + 63) / 64);
    gemm_wmma<<<grid, dim3(128), 0, s>>>(X, ldx, X2, ldx2, W, ldw, W2, ldw2,
                                         bias, Y, ldy, M, N, K, K2, act);
}

extern "C" void kernel_launch(void* const* d_in, const int* in_sizes, int n_in,
                              void* d_out, int out_size, void* d_ws, size_t ws_size,
                              hipStream_t stream)
{
    (void)in_sizes; (void)n_in; (void)out_size; (void)ws_size;

    // pytree leaf order: obs, h0, c0, then params dict keys sorted:
    // actor, critic, fill, log_std, lstm[0]{Wdt,Whh,Wih,bih}, lstm[1]{...},
    // market, mature, risk, runner
    const float* obs     = (const float*)d_in[0];
    const float* h0      = (const float*)d_in[1];
    const float* c0      = (const float*)d_in[2];
    const float* actW0   = (const float*)d_in[3];   // (256,770)
    const float* actb0   = (const float*)d_in[4];
    const float* actW1   = (const float*)d_in[5];   // (4,256)
    const float* actb1   = (const float*)d_in[6];
    const float* critW0  = (const float*)d_in[7];   // (256,512)
    const float* critb0  = (const float*)d_in[8];
    const float* critW1  = (const float*)d_in[9];   // (1,256)
    const float* critb1  = (const float*)d_in[10];
    const float* fillW   = (const float*)d_in[11];  // (16,512)
    const float* fillbp  = (const float*)d_in[12];
    const float* log_std = (const float*)d_in[13];  // (64,)
    const float* Wdt0    = (const float*)d_in[14];
    const float* Whh0    = (const float*)d_in[15];  // (2048,512)
    const float* Wih0    = (const float*)d_in[16];  // (2048,768)
    const float* bih0    = (const float*)d_in[17];
    const float* Wdt1    = (const float*)d_in[18];
    const float* Whh1    = (const float*)d_in[19];  // (2048,512)
    const float* Wih1    = (const float*)d_in[20];  // (2048,512)
    const float* bih1    = (const float*)d_in[21];
    const float* mW0     = (const float*)d_in[22];  // (256,56)
    const float* mb0     = (const float*)d_in[23];
    const float* mW1     = (const float*)d_in[24];
    const float* mb1     = (const float*)d_in[25];
    const float* mW2     = (const float*)d_in[26];
    const float* mb2     = (const float*)d_in[27];
    const float* matW    = (const float*)d_in[28];  // (16,512)
    const float* matbp   = (const float*)d_in[29];
    const float* riskW   = (const float*)d_in[30];  // (32,512)
    const float* riskbp  = (const float*)d_in[31];
    const float* rW0     = (const float*)d_in[32];  // (256,32)
    const float* rb0     = (const float*)d_in[33];
    const float* rW1     = (const float*)d_in[34];
    const float* rb1     = (const float*)d_in[35];
    const float* rW2     = (const float*)d_in[36];
    const float* rb2     = (const float*)d_in[37];

    float* ws = (float*)d_ws;
    size_t off = 0;
    float* Xr      = ws + off; off += (size_t)MR_ * 32;
    float* bufA    = ws + off; off += (size_t)MR_ * 256;  // runner ping / G1 / hidden
    float* bufB    = ws + off; off += (size_t)MR_ * 256;  // runner pong / r_emb
    float* mA      = ws + off; off += (size_t)BT_ * 256;
    float* mB      = ws + off; off += (size_t)BT_ * 256;
    float* lstm_in = ws + off; off += (size_t)BT_ * 768;
    float* lstm_o  = ws + off; off += (size_t)BT_ * 512;
    float* gates   = ws + off; off += (size_t)B_ * 2048;
    float* hbuf    = ws + off; off += (size_t)2 * B_ * LSTMH_;
    float* cbuf    = ws + off; off += (size_t)2 * B_ * LSTMH_;
    float* fillb   = ws + off; off += (size_t)BT_ * 16;
    float* matb    = ws + off; off += (size_t)BT_ * 16;
    float* riskb   = ws + off; off += (size_t)BT_ * 32;
    float* crith   = ws + off; off += (size_t)BT_ * 256;
    float* valb    = ws + off; off += (size_t)BT_;
    float* G2      = ws + off; off += (size_t)BT_ * 256;
    float* actb    = ws + off; off += (size_t)MR_ * 4;

    // ---- market MLP (obs rows, K=56, lda=568) ----
    gemm(stream, obs, OBSW_, mW0, 56,  mb0, mA, 256, BT_, 256, 56,  1);
    gemm(stream, mA,  256,   mW1, 256, mb1, mB, 256, BT_, 256, 256, 1);
    gemm(stream, mB,  256,   mW2, 256, mb2, mA, 256, BT_, 256, 256, 1);   // m_emb = mA

    // ---- runner MLP ----
    extract_runners<<<((size_t)MR_ * 32 + 255) / 256, 256, 0, stream>>>(obs, Xr);
    gemm(stream, Xr,   32,  rW0, 32,  rb0, bufB, 256, MR_, 256, 32,  1);
    gemm(stream, bufB, 256, rW1, 256, rb1, bufA, 256, MR_, 256, 256, 1);
    gemm(stream, bufA, 256, rW2, 256, rb2, bufB, 256, MR_, 256, 256, 1);  // r_emb = bufB

    // ---- lstm input: [m_emb | mean_r | max_r] ----
    build_lstm_in<<<((size_t)BT_ * 256 + 255) / 256, 256, 0, stream>>>(mA, bufB, lstm_in);

    // ---- init h, c ----
    hipMemcpyAsync(hbuf, h0, sizeof(float) * 2 * B_ * LSTMH_, hipMemcpyDeviceToDevice, stream);
    hipMemcpyAsync(cbuf, c0, sizeof(float) * 2 * B_ * LSTMH_, hipMemcpyDeviceToDevice, stream);

    float* h0b = hbuf;
    float* h1b = hbuf + B_ * LSTMH_;
    float* c0b = cbuf;
    float* c1b = cbuf + B_ * LSTMH_;

    // ---- LSTM over time (sequential); fused x/h gate GEMM per layer ----
    for (int t = 0; t < T_; ++t) {
        // layer 0: gates = x_t @ Wih0^T + h0 @ Whh0^T + bih0
        gemm_fused2(stream,
                    lstm_in + (size_t)t * 768, T_ * 768, Wih0, 768, 768,
                    h0b, LSTMH_, Whh0, LSTMH_, LSTMH_,
                    bih0, gates, 2048, B_, 2048, 0);
        lstm_pointwise<<<(B_ * LSTMH_ + 255) / 256, 256, 0, stream>>>(
            gates, Wdt0, obs, t, h0b, c0b, nullptr);
        // layer 1: gates = h0_new @ Wih1^T + h1 @ Whh1^T + bih1
        gemm_fused2(stream,
                    h0b, LSTMH_, Wih1, LSTMH_, LSTMH_,
                    h1b, LSTMH_, Whh1, LSTMH_, LSTMH_,
                    bih1, gates, 2048, B_, 2048, 0);
        lstm_pointwise<<<(B_ * LSTMH_ + 255) / 256, 256, 0, stream>>>(
            gates, Wdt1, obs, t, h1b, c1b, lstm_o);
    }

    // ---- heads on lstm_out ----
    gemm(stream, lstm_o, 512, fillW,  512, fillbp, fillb, 16,  BT_, 16,  512, 2);
    gemm(stream, lstm_o, 512, matW,   512, matbp,  matb,  16,  BT_, 16,  512, 2);
    gemm(stream, lstm_o, 512, riskW,  512, riskbp, riskb, 32,  BT_, 32,  512, 0);
    gemm(stream, lstm_o, 512, critW0, 512, critb0, crith, 256, BT_, 256, 512, 1);
    gemm(stream, crith,  256, critW1, 256, critb1, valb,  1,   BT_, 1,   256, 0);

    // ---- actor, factored: hidden = relu(r_emb@W1a^T + lstm@W1b^T + rank-1 + b) ----
    gemm(stream, lstm_o, 512, actW0 + 256, 770, nullptr, G2,   256, BT_, 256, 512, 0);
    gemm(stream, bufB,   256, actW0,       770, nullptr, bufA, 256, MR_, 256, 256, 0);
    actor_epilogue<<<((size_t)MR_ * 256 + 255) / 256, 256, 0, stream>>>(
        bufA, G2, fillb, matb, actW0, actb0);
    gemm(stream, bufA, 256, actW1, 256, actb1, actb, 4, MR_, 4, 256, 0);

    // ---- assemble final output ----
    assemble_out<<<(BT_ * 193 + 255) / 256, 256, 0, stream>>>(
        actb, log_std, valb, fillb, matb, riskb, (float*)d_out);
}
